// MyModel_32581621907558
// MI455X (gfx1250) — matrix operand order repaired and checked by reading, not compiled, not run on hardware
//
#include <hip/hip_runtime.h>
#include <hip/hip_bf16.h>

// ---------------------------------------------------------------------------
// 2-layer LSTM (H=51), B=512, T=2048 (+16 future) for MI455X / gfx1250.
//
// v3 changes vs v2 (driven by asm inspection):
//   * Gate dim padded to 256 -> exactly 4 N-tiles per wave, fully unrolled,
//     wave id scalarized via readfirstlane: GEMM loops are straight-line
//     uniform code (no exec-mask loops, no cndmask clamp, no reg rotation).
//   * Element-wise remapped to m=tid>>3, j=(tid&7)+8*it (it unrolled 0..6):
//     no integer division, all LDS accesses are base+immediate offsets,
//     bases hoisted out of the 2064-step loop.
//   * LDS section pointers marked __restrict__.
// ---------------------------------------------------------------------------

typedef _Float16 half_t;
typedef __attribute__((ext_vector_type(16))) _Float16 v16h;
typedef __attribute__((ext_vector_type(8)))  _Float16 v8h;
typedef __attribute__((ext_vector_type(8)))  float    v8f;
typedef __attribute__((ext_vector_type(4)))  float    v4f;

#define HID   51          // hidden size
#define G4    204         // 4*HID (real gate width)
#define GP    256         // gate width padded to 16 tiles (rows 204..255 zero)
#define NT4   4           // N-tiles per wave (16 tiles / 4 waves)
#define SW1   72          // W1 LDS row stride (halves), K padded to 64
#define SW2   136         // W2 LDS row stride (halves), K padded to 128
#define SG    260         // gate scratch row stride (floats), 1040B = 65*16
#define SH1   72          // h1pad row stride (halves)
#define SH12  136         // [h1|h2] row stride (halves)
#define SC    52          // c/h f32 row stride (floats), 208B = 13*16
#define ROWS  16          // batch rows per block
#define NBLK  32          // 512 / 16
#define NTHR  128         // 4 waves

// ---- LDS layout (bytes, every section 16B aligned) ----
#define LDS_W1    0
#define LDS_W2    (LDS_W1    + GP*SW1*2)     // 36864
#define LDS_B1    (LDS_W2    + GP*SW2*2)     // +69632
#define LDS_B2    (LDS_B1    + GP*4)
#define LDS_WX1   (LDS_B2    + GP*4)
#define LDS_WL    (LDS_WX1   + GP*4)
#define LDS_GATES (LDS_WL    + 64*4)
#define LDS_H1P   (LDS_GATES + ROWS*SG*4)
#define LDS_H12   (LDS_H1P   + ROWS*SH1*2)
#define LDS_C1    (LDS_H12   + ROWS*SH12*2)
#define LDS_C2    (LDS_C1    + ROWS*SC*4)
#define LDS_H2F   (LDS_C2    + ROWS*SC*4)
#define LDS_XB    (LDS_H2F   + ROWS*SC*4)
#define LDS_TOTAL (LDS_XB    + ROWS*4)       // = 143168 bytes (< 160KB)

union V16U { v16h v; v8h h2[2]; };
union V8FU { v8f  v; float f[8]; };

__device__ __forceinline__ float sigf(float x) {
    return __builtin_amdgcn_rcpf(1.0f + __expf(-x));       // saturates 0/1
}
__device__ __forceinline__ float tanhfast(float x) {
    return 1.0f - 2.0f * __builtin_amdgcn_rcpf(1.0f + __expf(2.0f * x)); // +-1
}

// A fragment (16-bit 16x32, ISA 7.12.2): lane m=l%16, half=l/16 holds
// K = 32*kk + 8*half + {0..7}  and  K = 32*kk + 16 + 8*half + {0..7}.
__device__ __forceinline__ v16h afrag(const half_t* A, int stride, int kk, int lane) {
    int m = lane & 15, hf = lane >> 4;
    const half_t* p = A + m * stride + kk * 32 + hf * 8;
    V16U r;
    r.h2[0] = *(const v8h*)p;
    r.h2[1] = *(const v8h*)(p + 16);
    return r.v;
}

// B fragment (32x16): lane holds N = 16*nt + l%16, K = 32*kk + 8*(l/16) + {0..7}
// and K = 32*kk + 16 + 8*(l/16) + {0..7}.  Weight stored row-major W[n][k] => B[k][n] = W[n][k].
__device__ __forceinline__ v16h bfrag(const half_t* W, int stride, int nt, int kk, int lane) {
    int n = nt * 16 + (lane & 15), hf = lane >> 4;
    const half_t* p = W + n * stride + kk * 32 + hf * 8;
    V16U r;
    r.h2[0] = *(const v8h*)p;
    r.h2[1] = *(const v8h*)(p + 16);
    return r.v;
}

__global__ __launch_bounds__(NTHR, 1)
void lstm2_wmma_kernel(const float* __restrict__ input,
                       const float* __restrict__ Wih1, const float* __restrict__ Whh1,
                       const float* __restrict__ bih1, const float* __restrict__ bhh1,
                       const float* __restrict__ Wih2, const float* __restrict__ Whh2,
                       const float* __restrict__ bih2, const float* __restrict__ bhh2,
                       const float* __restrict__ Wlin, const float* __restrict__ blin,
                       int T, int TT, float* __restrict__ out) {
    extern __shared__ char smem[];
    half_t* __restrict__ W1    = (half_t*)(smem + LDS_W1);
    half_t* __restrict__ W2    = (half_t*)(smem + LDS_W2);
    float*  __restrict__ b1i   = (float*) (smem + LDS_B1);   // permuted p=4j+g
    float*  __restrict__ b2i   = (float*) (smem + LDS_B2);
    float*  __restrict__ wx1i  = (float*) (smem + LDS_WX1);
    float*  __restrict__ wls   = (float*) (smem + LDS_WL);
    float*  __restrict__ gates = (float*) (smem + LDS_GATES);
    half_t* __restrict__ h1p   = (half_t*)(smem + LDS_H1P);
    half_t* __restrict__ h12   = (half_t*)(smem + LDS_H12);
    float*  __restrict__ c1s   = (float*) (smem + LDS_C1);
    float*  __restrict__ c2s   = (float*) (smem + LDS_C2);
    float*  __restrict__ h2f   = (float*) (smem + LDS_H2F);
    float*  __restrict__ xb    = (float*) (smem + LDS_XB);

    const int tid  = threadIdx.x;        // 0..127
    const int lane = tid & 31;           // lane within wave (WMMA layout)
    // wave id, forced into an SGPR so tile bases are compiler-uniform
    const int wvu  = __builtin_amdgcn_readfirstlane(tid >> 5);
    const int nt0  = wvu * NT4;          // this wave's first N-tile
    const int row0 = blockIdx.x * ROWS;

    // ---- one-time init: permuted f16 weights into LDS, zero state ----
    // Row permutation: new row p -> old gate row n = (p&3)*51 + (p>>2).
    for (int idx = tid; idx < GP * SW1; idx += NTHR) {
        int p = idx / SW1, k = idx - p * SW1;
        float v = 0.0f;
        if (p < G4 && k < HID) {
            int n = (p & 3) * HID + (p >> 2);
            v = Whh1[n * HID + k];
        }
        W1[idx] = (half_t)v;
    }
    for (int idx = tid; idx < GP * SW2; idx += NTHR) {
        int p = idx / SW2, k = idx - p * SW2;
        float v = 0.0f;
        if (p < G4) {
            int n = (p & 3) * HID + (p >> 2);
            if (k < HID)          v = Wih2[n * HID + k];
            else if (k < 2 * HID) v = Whh2[n * HID + (k - HID)];
        }
        W2[idx] = (half_t)v;
    }
    for (int p = tid; p < GP; p += NTHR) {
        int n = (p & 3) * HID + (p >> 2);
        bool ok = (p < G4);
        b1i[p]  = ok ? (bih1[n] + bhh1[n]) : 0.0f;
        b2i[p]  = ok ? (bih2[n] + bhh2[n]) : 0.0f;
        wx1i[p] = ok ? Wih1[n] : 0.0f;           // W_ih1 is (4H,1)
    }
    for (int j = tid; j < 64; j += NTHR) wls[j] = (j < HID) ? Wlin[j] : 0.0f;
    for (int i = tid; i < ROWS * SH1;  i += NTHR) h1p[i] = (half_t)0.0f;
    for (int i = tid; i < ROWS * SH12; i += NTHR) h12[i] = (half_t)0.0f;
    for (int i = tid; i < ROWS * SC;   i += NTHR) { c1s[i] = 0.0f; c2s[i] = 0.0f; h2f[i] = 0.0f; }
    if (tid < ROWS) xb[tid] = 0.0f;
    const float blin_v = blin[0];
    __syncthreads();

    // ---- hoisted bases ----
    // C-tile store base: row mh + r, col nt*16 + (lane&15)
    const int   mh  = (lane >> 4) * 8;
    float* __restrict__ gsp = gates + mh * SG + (lane & 15) + nt0 * 16;
    // element-wise mapping: m = tid>>3 (row), j = (tid&7) + 8*it
    const int em = tid >> 3, ej = tid & 7;
    float*  __restrict__ gep  = gates + em * SG + 4 * ej;
    const float* __restrict__ b1p = b1i  + 4 * ej;
    const float* __restrict__ b2p = b2i  + 4 * ej;
    const float* __restrict__ wxp = wx1i + 4 * ej;
    float*  __restrict__ c1p  = c1s + em * SC + ej;
    float*  __restrict__ c2p  = c2s + em * SC + ej;
    half_t* __restrict__ h1pp = h1p + em * SH1 + ej;
    half_t* __restrict__ h121 = h12 + em * SH12 + ej;
    half_t* __restrict__ h122 = h12 + em * SH12 + HID + ej;
    float*  __restrict__ h2fp = h2f + em * SC + ej;

    // preload x[t=0] (threads 0..15 own the scalar input pipeline)
    float xr = 0.0f;
    if (tid < ROWS) xr = input[(size_t)(row0 + tid) * T + 0];

    const v8f vzero = {0.f, 0.f, 0.f, 0.f, 0.f, 0.f, 0.f, 0.f};

    for (int t = 0; t < TT; ++t) {
        // publish current x; prefetch next step's x (hidden behind GEMM1)
        if (tid < ROWS) xb[tid] = xr;
        float xr_next = 0.0f;
        if ((t + 1) < T && tid < ROWS)
            xr_next = input[(size_t)(row0 + tid) * T + (t + 1)];

        // -------- GEMM1: gates = h1 @ Whh1'^T (K=64 pad), 4 tiles, unrolled -
        {
            v16h a0  = afrag(h1p, SH1, 0, lane);
            v16h a1  = afrag(h1p, SH1, 1, lane);
            v16h cb0 = bfrag(W1, SW1, nt0, 0, lane);
            v16h cb1 = bfrag(W1, SW1, nt0, 1, lane);
            #pragma unroll
            for (int i = 0; i < NT4; ++i) {
                v16h nb0 = cb0, nb1 = cb1;
                if (i + 1 < NT4) {                 // compile-time condition
                    nb0 = bfrag(W1, SW1, nt0 + i + 1, 0, lane);
                    nb1 = bfrag(W1, SW1, nt0 + i + 1, 1, lane);
                }
                V8FU acc;
                acc.v = __builtin_amdgcn_wmma_f32_16x16x32_f16(
                    false, a0, false, cb0, (short)0, vzero, false, false);
                acc.v = __builtin_amdgcn_wmma_f32_16x16x32_f16(
                    false, a1, false, cb1, (short)0, acc.v, false, false);
                #pragma unroll
                for (int r = 0; r < 8; ++r)
                    gsp[r * SG + i * 16] = acc.f[r];
                cb0 = nb0; cb1 = nb1;              // renamed away by unroll
            }
        }
        __syncthreads();   // B1: gates ready

        // -------- element-wise LSTM cell 1 (gates interleaved 4j..4j+3) -----
        {
            float x = xb[em];
            #pragma unroll
            for (int it = 0; it < 7; ++it) {
                if (ej + 8 * it < HID) {           // static for it<6
                    v4f g = *(const v4f*)(gep + 32 * it);
                    v4f b = *(const v4f*)(b1p + 32 * it);
                    v4f w = *(const v4f*)(wxp + 32 * it);
                    float gi = g[0] + b[0] + x * w[0];
                    float gf = g[1] + b[1] + x * w[1];
                    float gg = g[2] + b[2] + x * w[2];
                    float go = g[3] + b[3] + x * w[3];
                    float c  = sigf(gf) * c1p[8 * it] + sigf(gi) * tanhfast(gg);
                    c1p[8 * it] = c;
                    float h  = sigf(go) * tanhfast(c);
                    h1pp[8 * it] = (half_t)h;      // A of next step's GEMM1
                    h121[8 * it] = (half_t)h;      // K 0..50 of GEMM2's A
                }
            }
        }
        __syncthreads();   // B2: h1 ready

        // -------- GEMM2: gates = [h1|h2] @ [Wih2|Whh2]'^T (K=128 pad) -------
        {
            v16h a[4];
            #pragma unroll
            for (int kk = 0; kk < 4; ++kk) a[kk] = afrag(h12, SH12, kk, lane);
            v16h cb[4];
            #pragma unroll
            for (int kk = 0; kk < 4; ++kk) cb[kk] = bfrag(W2, SW2, nt0, kk, lane);
            #pragma unroll
            for (int i = 0; i < NT4; ++i) {
                v16h nb[4];
                #pragma unroll
                for (int kk = 0; kk < 4; ++kk) nb[kk] = cb[kk];
                if (i + 1 < NT4) {
                    #pragma unroll
                    for (int kk = 0; kk < 4; ++kk)
                        nb[kk] = bfrag(W2, SW2, nt0 + i + 1, kk, lane);
                }
                V8FU acc; acc.v = vzero;
                #pragma unroll
                for (int kk = 0; kk < 4; ++kk)
                    acc.v = __builtin_amdgcn_wmma_f32_16x16x32_f16(
                        false, a[kk], false, cb[kk], (short)0, acc.v, false, false);
                #pragma unroll
                for (int r = 0; r < 8; ++r)
                    gsp[r * SG + i * 16] = acc.f[r];
                #pragma unroll
                for (int kk = 0; kk < 4; ++kk) cb[kk] = nb[kk];
            }
        }
        __syncthreads();   // B3: gates ready

        // -------- element-wise LSTM cell 2 ----------------------------------
        {
            #pragma unroll
            for (int it = 0; it < 7; ++it) {
                if (ej + 8 * it < HID) {
                    v4f g = *(const v4f*)(gep + 32 * it);
                    v4f b = *(const v4f*)(b2p + 32 * it);
                    float gi = g[0] + b[0];
                    float gf = g[1] + b[1];
                    float gg = g[2] + b[2];
                    float go = g[3] + b[3];
                    float c  = sigf(gf) * c2p[8 * it] + sigf(gi) * tanhfast(gg);
                    c2p[8 * it] = c;
                    float h  = sigf(go) * tanhfast(c);
                    h122[8 * it] = (half_t)h;      // K 51..101 of GEMM2's A
                    h2fp[8 * it] = h;              // f32 copy for output dot
                }
            }
        }
        __syncthreads();   // B4: h2 ready

        // -------- linear head (float4 dot) + autoregressive feedback --------
        if (tid < ROWS) {
            float o = blin_v;
            #pragma unroll
            for (int jj = 0; jj < 13; ++jj) {
                v4f hv  = *(const v4f*)(h2f + tid * SC + 4 * jj);
                v4f wv4 = *(const v4f*)(wls + 4 * jj);
                o += hv[0] * wv4[0] + hv[1] * wv4[1] + hv[2] * wv4[2] + hv[3] * wv4[3];
            }
            out[(size_t)(row0 + tid) * TT + t] = o;
            if ((t + 1) >= T) xr_next = o;        // feed output back as x
        }
        xr = xr_next;
    }
}

extern "C" void kernel_launch(void* const* d_in, const int* in_sizes, int n_in,
                              void* d_out, int out_size, void* d_ws, size_t ws_size,
                              hipStream_t stream) {
    const float* input = (const float*)d_in[0];
    const float* Wih1  = (const float*)d_in[1];
    const float* Whh1  = (const float*)d_in[2];
    const float* bih1  = (const float*)d_in[3];
    const float* bhh1  = (const float*)d_in[4];
    const float* Wih2  = (const float*)d_in[5];
    const float* Whh2  = (const float*)d_in[6];
    const float* bih2  = (const float*)d_in[7];
    const float* bhh2  = (const float*)d_in[8];
    const float* Wlin  = (const float*)d_in[9];
    const float* blin  = (const float*)d_in[10];
    // d_in[11] is `future` on device; derive host-side instead:
    const int B  = 512;
    const int T  = in_sizes[0] / B;   // 2048
    const int TT = out_size / B;      // T + future

    (void)n_in; (void)d_ws; (void)ws_size;

    hipFuncSetAttribute((const void*)lstm2_wmma_kernel,
                        hipFuncAttributeMaxDynamicSharedMemorySize, LDS_TOTAL);

    lstm2_wmma_kernel<<<dim3(NBLK), dim3(NTHR), LDS_TOTAL, stream>>>(
        input, Wih1, Whh1, bih1, bhh1, Wih2, Whh2, bih2, bhh2,
        Wlin, blin, T, TT, (float*)d_out);
}
